// EdgeScoringNetwork_21345987461746
// MI455X (gfx1250) — compile-verified, matrix operand order checked by reading, not run on hardware
//
#include <hip/hip_runtime.h>
#include <hip/hip_bf16.h>
#include <math.h>

// ---------------------------------------------------------------------------
// EdgeScoringNetwork on gfx1250: L2-normalize -> gather -> MLP(257,128,128,1)
// -> hard-concrete gate.  GEMMs run on v_wmma_f32_16x16x32_f16 (one 16-edge
// tile per wave32).  Compute-bound: ~64 GFLOP vs ~40 MB of HBM traffic.
// ---------------------------------------------------------------------------

typedef __attribute__((ext_vector_type(16))) _Float16 v16h;
typedef __attribute__((ext_vector_type(8)))  _Float16 v8h;
typedef __attribute__((ext_vector_type(4)))  _Float16 v4h;
typedef __attribute__((ext_vector_type(8)))  float    v8f;

#define N_NODES   50000
#define N_EDGES   600000
#define D_IN      128
#define W1_K      288      // 257 padded to 9 chunks of K=32 (pad rows are zero)
#define H1_STRIDE 136      // padded LDS row stride (f16 elems) for h1 transpose
#define N_TILES   (N_EDGES / 16)   // 37500 exactly

// Load one lane's slice of a 16x32 f16 A/B fragment (K contiguous in memory).
// Lane L: rows/cols index = L&15, K = kbase + (L>>4)*8 + [0..7] and +16.
// v16h element j<8  -> VGPR j/2 half j%2 -> K = grp*8 + j      (per ISA table)
// v16h element j>=8 -> VGPR 4+(j-8)/2    -> K = 16 + grp*8 + (j-8)
__device__ __forceinline__ v16h load_frag16(const _Float16* base) {
  v8h lo = *(const v8h*)(base);        // 16B -> global/ds _b128
  v8h hi = *(const v8h*)(base + 16);
  return __builtin_shufflevector(lo, hi, 0,1,2,3,4,5,6,7,8,9,10,11,12,13,14,15);
}

// ---------------------------------------------------------------------------
// Kernel 1: L2-normalize node features, emit f16 table (halves gather bytes).
// One wave per node: 4 floats/lane, 5-step shuffle reduction.
// ---------------------------------------------------------------------------
__global__ void __launch_bounds__(256)
normalize_nodes(const float* __restrict__ nf_in, _Float16* __restrict__ nf_out)
{
  const int node = blockIdx.x * 8 + (threadIdx.x >> 5);
  if (node >= N_NODES) return;
  const int lane = threadIdx.x & 31;
  const float4 v = *(const float4*)(nf_in + (size_t)node * D_IN + lane * 4);
  float ss = v.x*v.x + v.y*v.y + v.z*v.z + v.w*v.w;
  ss += __shfl_xor(ss, 1, 32);
  ss += __shfl_xor(ss, 2, 32);
  ss += __shfl_xor(ss, 4, 32);
  ss += __shfl_xor(ss, 8, 32);
  ss += __shfl_xor(ss, 16, 32);
  const float scale = 1.0f / (sqrtf(ss) + 1e-8f);
  v4h o;
  o[0] = (_Float16)(v.x * scale);
  o[1] = (_Float16)(v.y * scale);
  o[2] = (_Float16)(v.z * scale);
  o[3] = (_Float16)(v.w * scale);
  *(v4h*)(nf_out + (size_t)node * D_IN + lane * 4) = o;
}

// ---------------------------------------------------------------------------
// Kernel 2: cast weights to f16; pad W1's K dim 257 -> 288 with zeros so the
// 9th WMMA K-chunk (holding only the dist column at K=256) is exact.
// ---------------------------------------------------------------------------
__global__ void __launch_bounds__(256)
pack_weights(const float* __restrict__ W1, const float* __restrict__ W2,
             const float* __restrict__ W3,
             _Float16* __restrict__ W1h, _Float16* __restrict__ W2h,
             _Float16* __restrict__ W3h)
{
  const int i = blockIdx.x * 256 + threadIdx.x;
  if (i < 128 * W1_K) {
    const int n = i / W1_K, k = i % W1_K;
    W1h[i] = (k < 257) ? (_Float16)W1[n * 257 + k] : (_Float16)0.0f;
  }
  if (i < 128 * 128) W2h[i] = (_Float16)W2[i];
  if (i < 128)       W3h[i] = (_Float16)W3[i];
}

// ---------------------------------------------------------------------------
// Kernel 3: per-edge MLP on WMMA.  One wave = one 16-edge tile.
// ---------------------------------------------------------------------------
__global__ void __launch_bounds__(256)
edge_mlp_wmma(const _Float16* __restrict__ nf,
              const int* __restrict__ ei,          // [2][E] int32
              const _Float16* __restrict__ W1h, const float* __restrict__ b1,
              const _Float16* __restrict__ W2h, const float* __restrict__ b2,
              const _Float16* __restrict__ W3h, const float* __restrict__ b3,
              float* __restrict__ out_w, float* __restrict__ out_la)
{
  __shared__ __align__(16) _Float16 h1s[8][16][H1_STRIDE];  // 34816 B

  const int lane = threadIdx.x & 31;
  const int wave = threadIdx.x >> 5;
  const int tile = blockIdx.x * 8 + wave;
  if (tile >= N_TILES) return;          // whole-wave exit: EXEC stays all-1s

  const int m   = lane & 15;            // edge row this lane serves in A-frags
  const int grp = lane >> 4;            // lane half -> K sub-range
  const int e0  = tile * 16;

  const int si = ei[e0 + m];
  const int ti = ei[N_EDGES + e0 + m];
  const _Float16* hsrow = nf + (size_t)si * D_IN;
  const _Float16* htrow = nf + (size_t)ti * D_IN;

  // ---- gather layer-1 A fragments: chunks 0-3 = hs, 4-7 = ht -------------
  v16h A[9];
  #pragma unroll
  for (int c = 0; c < 4; ++c) A[c]     = load_frag16(hsrow + 32 * c + grp * 8);
  #pragma unroll
  for (int c = 0; c < 4; ++c) A[4 + c] = load_frag16(htrow + 32 * c + grp * 8);

  // ---- dist = ||hs - ht|| from already-loaded fragments ------------------
  // lanes L and L^16 hold disjoint K-subsets of the same edge row m.
  float dsum = 0.0f;
  #pragma unroll
  for (int c = 0; c < 4; ++c)
    #pragma unroll
    for (int j = 0; j < 16; ++j) {
      const float d = (float)A[c][j] - (float)A[4 + c][j];
      dsum = fmaf(d, d, dsum);
    }
  dsum += __shfl_xor(dsum, 16, 32);
  const float dist = sqrtf(dsum);

  // chunk 8: only K=256 is live -> element 0 of group-0 lanes
  v16h a8 = {};
  a8[0] = (grp == 0) ? (_Float16)dist : (_Float16)0.0f;
  A[8] = a8;

  // ---- layer 1: [16x288] @ [288x128], 72 WMMAs ---------------------------
  v8f acc[8];
  #pragma unroll
  for (int nt = 0; nt < 8; ++nt) acc[nt] = {};
  #pragma unroll
  for (int c = 0; c < 9; ++c) {
    #pragma unroll
    for (int nt = 0; nt < 8; ++nt) {
      const int n = nt * 16 + m;        // B column held by this lane
      const v16h b = load_frag16(W1h + (size_t)n * W1_K + 32 * c + grp * 8);
      acc[nt] = __builtin_amdgcn_wmma_f32_16x16x32_f16(
          false, A[c], false, b, (short)0, acc[nt], false, false);
    }
  }

  // ---- bias + ReLU, stage h1 through LDS (D-layout -> A-layout transpose)
  #pragma unroll
  for (int nt = 0; nt < 8; ++nt) {
    const int n = nt * 16 + m;
    const float bias = b1[n];
    #pragma unroll
    for (int v = 0; v < 8; ++v) {
      const float h = fmaxf(acc[nt][v] + bias, 0.0f);
      h1s[wave][v + 8 * grp][n] = (_Float16)h;   // D: lane=col n, VGPR v=row
    }
  }
  asm volatile("s_wait_dscnt 0" ::: "memory");   // wave-private region: no bar

  v16h A2[4];
  #pragma unroll
  for (int c = 0; c < 4; ++c)
    A2[c] = load_frag16(&h1s[wave][m][32 * c + grp * 8]);

  // ---- layer 2: [16x128] @ [128x128], 32 WMMAs ---------------------------
  #pragma unroll
  for (int nt = 0; nt < 8; ++nt) acc[nt] = {};
  #pragma unroll
  for (int c = 0; c < 4; ++c) {
    #pragma unroll
    for (int nt = 0; nt < 8; ++nt) {
      const int n = nt * 16 + m;
      const v16h b = load_frag16(W2h + (size_t)n * D_IN + 32 * c + grp * 8);
      acc[nt] = __builtin_amdgcn_wmma_f32_16x16x32_f16(
          false, A2[c], false, b, (short)0, acc[nt], false, false);
    }
  }

  // ---- layer 3 (128-dot with W3) + hard-concrete gate --------------------
  // D layout: lane holds column n = nt*16 + m, VGPR v holds row v + 8*grp.
  float pv[8];
  #pragma unroll
  for (int v = 0; v < 8; ++v) pv[v] = 0.0f;
  #pragma unroll
  for (int nt = 0; nt < 8; ++nt) {
    const int n = nt * 16 + m;
    const float bias = b2[n];
    const float w3n  = (float)W3h[n];
    #pragma unroll
    for (int v = 0; v < 8; ++v) {
      const float h = fmaxf(acc[nt][v] + bias, 0.0f);
      pv[v] = fmaf(h, w3n, pv[v]);
    }
  }
  const float b3v = b3[0];
  #pragma unroll
  for (int v = 0; v < 8; ++v) {
    float s = pv[v];
    s += __shfl_xor(s, 1, 32);          // reduce within each 16-lane half:
    s += __shfl_xor(s, 2, 32);          // half 0 -> edge row v,
    s += __shfl_xor(s, 4, 32);          // half 1 -> edge row v + 8
    s += __shfl_xor(s, 8, 32);
    if (m == 0) {                       // lanes 0 and 16 emit
      const float la  = s + b3v;
      const int   row = v + 8 * grp;
      out_la[e0 + row] = la;
      const float sg = 1.0f / (1.0f + __expf(-la));
      float w = fmaf(sg, 1.2f, -0.1f);  // sigmoid*(zeta-gamma)+gamma
      out_w[e0 + row] = fminf(fmaxf(w, 0.0f), 1.0f);
    }
  }
}

// ---------------------------------------------------------------------------
extern "C" void kernel_launch(void* const* d_in, const int* in_sizes, int n_in,
                              void* d_out, int out_size, void* d_ws, size_t ws_size,
                              hipStream_t stream)
{
  (void)in_sizes; (void)n_in; (void)out_size; (void)ws_size;

  const float* node_feat = (const float*)d_in[0];
  const int*   ei        = (const int*)d_in[1];    // jnp.int64 sans x64 -> int32
  const float* W1 = (const float*)d_in[2];
  const float* b1 = (const float*)d_in[3];
  const float* W2 = (const float*)d_in[4];
  const float* b2 = (const float*)d_in[5];
  const float* W3 = (const float*)d_in[6];
  const float* b3 = (const float*)d_in[7];

  float* out_w  = (float*)d_out;             // edge_weights [E]
  float* out_la = (float*)d_out + N_EDGES;   // logAlpha     [E]

  // workspace layout (all regions rewritten every call -> deterministic)
  char*  ws  = (char*)d_ws;
  size_t off = 0;
  _Float16* nf_h = (_Float16*)(ws + off);
  off += (size_t)N_NODES * D_IN * sizeof(_Float16);    // 12.8 MB
  off  = (off + 255) & ~(size_t)255;
  _Float16* W1h = (_Float16*)(ws + off);
  off += (size_t)128 * W1_K * sizeof(_Float16);        // 73.7 KB
  off  = (off + 255) & ~(size_t)255;
  _Float16* W2h = (_Float16*)(ws + off);
  off += (size_t)128 * 128 * sizeof(_Float16);         // 32.8 KB
  off  = (off + 255) & ~(size_t)255;
  _Float16* W3h = (_Float16*)(ws + off);

  normalize_nodes<<<(N_NODES + 7) / 8, 256, 0, stream>>>(node_feat, nf_h);
  pack_weights<<<(128 * W1_K + 255) / 256, 256, 0, stream>>>(W1, W2, W3,
                                                             W1h, W2h, W3h);
  edge_mlp_wmma<<<(N_TILES + 7) / 8, 256, 0, stream>>>(
      nf_h, ei, W1h, b1, W2h, b2, W3h, b3, out_w, out_la);
}
